// MultiHeadAttention_14791867367560
// MI455X (gfx1250) — matrix operand order repaired and checked by reading, not compiled, not run on hardware
//
#include <hip/hip_runtime.h>

#define BATCH 2
#define SEQ   2048
#define HID   1024
#define NH    16
#define HD    64

typedef __attribute__((ext_vector_type(16))) __bf16 v16bf;
typedef __attribute__((ext_vector_type(8)))  float  v8f;
typedef __attribute__((ext_vector_type(4)))  int    v4i;

// 32-byte fragment: 16 bf16 elements, filled via two uint4 LDS/global loads.
union Frag32B {
    struct { uint4 lo, hi; } q;
    unsigned short us[16];
    v16bf v;
};

__device__ __forceinline__ unsigned short f2bf(float f) {
    unsigned int u = __float_as_uint(f);
    u += 0x7FFFu + ((u >> 16) & 1u);   // round-to-nearest-even
    return (unsigned short)(u >> 16);
}

// CDNA5 async global->LDS copy (ASYNCcnt-tracked), with staged fallback.
// Probe result (round 2 diagnostic): parameters are int4* in addrspace(1)/(3).
#if defined(__AMDGCN__) && __has_builtin(__builtin_amdgcn_global_load_async_to_lds_b128)
#define USE_ASYNC_LDS 1
typedef __attribute__((address_space(1))) void* as1p;
typedef __attribute__((address_space(3))) void* as3p;
typedef __attribute__((address_space(1))) v4i*  as1_v4i;
typedef __attribute__((address_space(3))) v4i*  as3_v4i;
#define GLDS_B128(gp, lp)                                         \
    __builtin_amdgcn_global_load_async_to_lds_b128(               \
        (as1_v4i)(as1p)(void*)(gp), (as3_v4i)(as3p)(void*)(lp), 0, 0)
#else
#define USE_ASYNC_LDS 0
#endif

__device__ __forceinline__ void wait_async0() {
#if __has_builtin(__builtin_amdgcn_s_wait_asynccnt)
    __builtin_amdgcn_s_wait_asynccnt(0);
#else
    asm volatile("s_wait_asynccnt 0" ::: "memory");
#endif
}

// ---------------------------------------------------------------------------
// Kernel 1: fp32 -> bf16 conversion (memory-bound, trivial)
// ---------------------------------------------------------------------------
__global__ void k_f32_to_bf16(const float* __restrict__ src,
                              unsigned short* __restrict__ dst, int n) {
    int i = blockIdx.x * blockDim.x + threadIdx.x;
    if (i < n) dst[i] = f2bf(src[i]);
}

// ---------------------------------------------------------------------------
// Kernel 2: bf16 GEMM  C[M][N] = A[M][K] * Bw[N][K]^T   (i.e. y = x @ W^T)
// 128x128x32 block tile, 8 waves, wave tile 64x32 (4x2 WMMA C tiles).
// Global->LDS via async-to-LDS (no VGPR staging) when available.
// ---------------------------------------------------------------------------
#define BM 128
#define BN 128
#define BK 32
#define LDT 40   // LDS tile row stride in ushorts (32 + 8 pad, 80B = 16B-mult)

__global__ __launch_bounds__(256) void k_gemm_bf16(
        const unsigned short* __restrict__ A,   // [M][K] bf16
        const unsigned short* __restrict__ Bw,  // [N][K] bf16
        float* __restrict__ C,                  // [M][N] fp32
        int M, int N, int K)
{
    __shared__ unsigned short As[2][BM * LDT];
    __shared__ unsigned short Bs[2][BN * LDT];

    const int tid   = threadIdx.x;
    const int lane  = tid & 31;
    const int wave  = tid >> 5;
    const int waveM = wave & 1;          // 2 waves along M (64 rows each)
    const int waveN = wave >> 1;         // 4 waves along N (32 cols each)
    const int half  = (lane >> 4) & 1;
    const int l16   = lane & 15;

    const int m0 = blockIdx.y * BM;
    const int n0 = blockIdx.x * BN;
    const int KT = K / BK;

    // Per-thread copy slots: rows (tid>>2) and (tid>>2)+64, 16B column (tid&3)*8.
    const int row0 = tid >> 2;
    const int row1 = row0 + 64;
    const int cv   = (tid & 3) * 8;

    const unsigned short* gA0 = A  + (size_t)(m0 + row0) * K + cv;
    const unsigned short* gA1 = A  + (size_t)(m0 + row1) * K + cv;
    const unsigned short* gB0 = Bw + (size_t)(n0 + row0) * K + cv;
    const unsigned short* gB1 = Bw + (size_t)(n0 + row1) * K + cv;
    const int sA0 = row0 * LDT + cv;
    const int sA1 = row1 * LDT + cv;

    v8f acc[4][2] = {};

#if USE_ASYNC_LDS
    // ---- async copy pipeline: memory -> LDS without touching VGPRs ----
    GLDS_B128(gA0, &As[0][sA0]);
    GLDS_B128(gA1, &As[0][sA1]);
    GLDS_B128(gB0, &Bs[0][sA0]);
    GLDS_B128(gB1, &Bs[0][sA1]);
    wait_async0();
    __syncthreads();
#else
    uint4 ra0 = *(const uint4*)(gA0);
    uint4 ra1 = *(const uint4*)(gA1);
    uint4 rb0 = *(const uint4*)(gB0);
    uint4 rb1 = *(const uint4*)(gB1);
    *(uint4*)(&As[0][sA0]) = ra0;
    *(uint4*)(&As[0][sA1]) = ra1;
    *(uint4*)(&Bs[0][sA0]) = rb0;
    *(uint4*)(&Bs[0][sA1]) = rb1;
    __syncthreads();
#endif

    for (int kt = 0; kt < KT; ++kt) {
        const int cur = kt & 1, nxt = cur ^ 1;
        const int koff = (kt + 1) * BK;

#if USE_ASYNC_LDS
        if (kt + 1 < KT) {   // start next tile's DMA before compute
            GLDS_B128(gA0 + koff, &As[nxt][sA0]);
            GLDS_B128(gA1 + koff, &As[nxt][sA1]);
            GLDS_B128(gB0 + koff, &Bs[nxt][sA0]);
            GLDS_B128(gB1 + koff, &Bs[nxt][sA1]);
        }
#else
        if (kt + 1 < KT) {
            ra0 = *(const uint4*)(gA0 + koff);
            ra1 = *(const uint4*)(gA1 + koff);
            rb0 = *(const uint4*)(gB0 + koff);
            rb1 = *(const uint4*)(gB1 + koff);
        }
#endif
        if (kt + 2 < KT) {   // L2 prefetch of tile after next
            __builtin_prefetch(gA0 + (kt + 2) * BK, 0, 0);
            __builtin_prefetch(gB0 + (kt + 2) * BK, 0, 0);
        }

        // A fragments: row = lane%16; lane<16 -> K {0-7,16-23}, lane>=16 -> {8-15,24-31}
        Frag32B af[4], bfg[2];
        #pragma unroll
        for (int t = 0; t < 4; ++t) {
            const unsigned short* p = &As[cur][(waveM * 64 + t * 16 + l16) * LDT];
            af[t].q.lo = *(const uint4*)(p + half * 8);
            af[t].q.hi = *(const uint4*)(p + 16 + half * 8);
        }
        // B fragments: N = lane%16; lane<16 -> K {0-7,16-23}, lane>=16 -> {8-15,24-31}
        #pragma unroll
        for (int u = 0; u < 2; ++u) {
            const unsigned short* p = &Bs[cur][(waveN * 32 + u * 16 + l16) * LDT];
            bfg[u].q.lo = *(const uint4*)(p + half * 8);
            bfg[u].q.hi = *(const uint4*)(p + 16 + half * 8);
        }
        #pragma unroll
        for (int t = 0; t < 4; ++t)
            #pragma unroll
            for (int u = 0; u < 2; ++u)
                acc[t][u] = __builtin_amdgcn_wmma_f32_16x16x32_bf16(
                    false, af[t].v, false, bfg[u].v, (short)0, acc[t][u], false, false);

#if USE_ASYNC_LDS
        if (kt + 1 < KT) wait_async0();
#else
        if (kt + 1 < KT) {
            *(uint4*)(&As[nxt][sA0]) = ra0;
            *(uint4*)(&As[nxt][sA1]) = ra1;
            *(uint4*)(&Bs[nxt][sA0]) = rb0;
            *(uint4*)(&Bs[nxt][sA1]) = rb1;
        }
#endif
        __syncthreads();
    }

    // C layout: VGPR r, lane -> row = r + 8*half, col = lane%16
    #pragma unroll
    for (int t = 0; t < 4; ++t)
        #pragma unroll
        for (int u = 0; u < 2; ++u) {
            int col = n0 + waveN * 32 + u * 16 + l16;
            #pragma unroll
            for (int r = 0; r < 8; ++r) {
                int row = m0 + waveM * 64 + t * 16 + half * 8 + r;
                C[(size_t)row * N + col] = acc[t][u][r];
            }
        }
}

// ---------------------------------------------------------------------------
// Kernel 3: RoPE (fp32) + repack q,k,v -> [B][NH][S][HD] bf16
// one thread per (b,h,s,dpair), dpair in [0,32)
// ---------------------------------------------------------------------------
__global__ void k_rope_pack(const float* __restrict__ qf,
                            const float* __restrict__ kf,
                            const float* __restrict__ vf,
                            unsigned short* __restrict__ qb,
                            unsigned short* __restrict__ kb,
                            unsigned short* __restrict__ vb)
{
    long i = (long)blockIdx.x * blockDim.x + threadIdx.x;
    const long total = (long)BATCH * NH * SEQ * (HD / 2);
    if (i >= total) return;
    int dp = (int)(i & 31);
    long r = i >> 5;
    int s = (int)(r % SEQ); r /= SEQ;
    int h = (int)(r % NH);
    int b = (int)(r / NH);

    // inv_freq = theta^(-2*dp/HD);  ln(10000) = 9.2103403...
    float inv = __expf(-(2.0f * dp / (float)HD) * 9.210340371976184f);
    float ang = (float)s * inv;
    float sn, cs;
    __sincosf(ang, &sn, &cs);

    size_t src = ((size_t)(b * SEQ + s)) * HID + (size_t)h * HD;
    size_t dst = (((size_t)(b * NH + h)) * SEQ + s) * HD;

    float q1 = qf[src + dp], q2 = qf[src + dp + 32];
    qb[dst + dp]      = f2bf(q1 * cs - q2 * sn);
    qb[dst + dp + 32] = f2bf(q2 * cs + q1 * sn);
    float k1 = kf[src + dp], k2 = kf[src + dp + 32];
    kb[dst + dp]      = f2bf(k1 * cs - k2 * sn);
    kb[dst + dp + 32] = f2bf(k2 * cs + k1 * sn);
    vb[dst + dp]      = f2bf(vf[src + dp]);
    vb[dst + dp + 32] = f2bf(vf[src + dp + 32]);
}

// ---------------------------------------------------------------------------
// Kernel 4: causal flash attention, bf16 WMMA, fp32 online softmax.
// block = 128 threads (4 waves), 64 Q rows per block, 16 per wave,
// KV processed in 32-column steps staged through LDS.
// ---------------------------------------------------------------------------
#define FL_LDK 72   // K-tile row stride (64 + 8), 144B = 16B-mult
#define FL_LDV 40   // V^T row stride   (32 + 8),  80B = 16B-mult
#define FL_LDP 40   // P scratch row stride

__global__ __launch_bounds__(128) void k_flash(
        const unsigned short* __restrict__ qb,
        const unsigned short* __restrict__ kb,
        const unsigned short* __restrict__ vb,
        unsigned short* __restrict__ attnb)
{
    __shared__ unsigned short Ks[32 * FL_LDK];     // K tile  [j][d]
    __shared__ unsigned short Vt[64 * FL_LDV];     // V^T     [d][j]
    __shared__ unsigned short Ps[4][16 * FL_LDP];  // per-wave P scratch

    const int tid  = threadIdx.x;
    const int lane = tid & 31;
    const int wave = tid >> 5;
    const int half = (lane >> 4) & 1;
    const int l16  = lane & 15;

    const int qt = blockIdx.x;
    const int h  = blockIdx.y;
    const int b  = blockIdx.z;

    const size_t bh = (size_t)(b * NH + h);
    const unsigned short* Qp = qb + bh * SEQ * HD;
    const unsigned short* Kp = kb + bh * SEQ * HD;
    const unsigned short* Vp = vb + bh * SEQ * HD;

    const int q0w  = qt * 64 + wave * 16;
    const int qrow = q0w + l16;

    // Preload Q A-fragments for both 32-wide K-chunks of HD=64.
    Frag32B qfr[2];
    #pragma unroll
    for (int c = 0; c < 2; ++c) {
        const unsigned short* p = Qp + (size_t)qrow * HD + c * 32;
        qfr[c].q.lo = *(const uint4*)(p + half * 8);
        qfr[c].q.hi = *(const uint4*)(p + 16 + half * 8);
    }

    v8f o[4] = {};
    float mrow[8], lrow[8];
    #pragma unroll
    for (int r = 0; r < 8; ++r) { mrow[r] = -1e30f; lrow[r] = 0.0f; }

    const float scale  = 0.125f;       // 1/sqrt(64)
    const int   nsteps = qt * 2 + 2;   // j0 = step*32 <= qt*64+63

    for (int step = 0; step < nsteps; ++step) {
        const int j0 = step * 32;
        __syncthreads();
        // Cooperative stage: K tile [32][64] direct, V transposed into Vt[d][j].
        #pragma unroll
        for (int i = 0; i < 2; ++i) {
            int idx = tid + i * 128;
            int row = idx >> 3, cvv = idx & 7;
            uint4 kd = *(const uint4*)(Kp + (size_t)(j0 + row) * HD + cvv * 8);
            *(uint4*)(&Ks[row * FL_LDK + cvv * 8]) = kd;
            uint4 vd = *(const uint4*)(Vp + (size_t)(j0 + row) * HD + cvv * 8);
            const unsigned short* ve = (const unsigned short*)&vd;
            #pragma unroll
            for (int e = 0; e < 8; ++e)
                Vt[(cvv * 8 + e) * FL_LDV + row] = ve[e];
        }
        __syncthreads();

        // Scores: S = Q K^T, two 16-column subtiles, 2 WMMAs each over d.
        v8f st[2];
        #pragma unroll
        for (int sub = 0; sub < 2; ++sub) {
            v8f acc = {};
            int jl = sub * 16 + l16;
            #pragma unroll
            for (int c = 0; c < 2; ++c) {
                Frag32B kfg;
                const unsigned short* p = &Ks[jl * FL_LDK + c * 32];
                kfg.q.lo = *(const uint4*)(p + half * 8);
                kfg.q.hi = *(const uint4*)(p + 16 + half * 8);
                acc = __builtin_amdgcn_wmma_f32_16x16x32_bf16(
                    false, qfr[c].v, false, kfg.v, (short)0, acc, false, false);
            }
            int jg = j0 + sub * 16 + l16;
            #pragma unroll
            for (int r = 0; r < 8; ++r) {
                int qg = q0w + half * 8 + r;
                float s = acc[r] * scale;
                acc[r] = (jg > qg) ? -1e30f : s;
            }
            st[sub] = acc;
        }

        // Online softmax: rows live across 16 lanes of one half-wave.
        float p0[8], p1[8];
        #pragma unroll
        for (int r = 0; r < 8; ++r) {
            float x = fmaxf(st[0][r], st[1][r]);
            #pragma unroll
            for (int off = 1; off < 16; off <<= 1)
                x = fmaxf(x, __shfl_xor(x, off, 32));
            float mnew  = fmaxf(mrow[r], x);
            float alpha = __expf(mrow[r] - mnew);
            float a0 = __expf(st[0][r] - mnew);
            float a1 = __expf(st[1][r] - mnew);
            float sum = a0 + a1;
            #pragma unroll
            for (int off = 1; off < 16; off <<= 1)
                sum += __shfl_xor(sum, off, 32);
            lrow[r] = lrow[r] * alpha + sum;
            mrow[r] = mnew;
            p0[r] = a0; p1[r] = a1;
            #pragma unroll
            for (int t = 0; t < 4; ++t) o[t][r] *= alpha;
        }

        // P: C-layout -> A-fragment layout via per-wave LDS scratch.
        unsigned short* ps = Ps[wave];
        #pragma unroll
        for (int r = 0; r < 8; ++r) {
            int row = half * 8 + r;
            ps[row * FL_LDP + l16]      = f2bf(p0[r]);
            ps[row * FL_LDP + 16 + l16] = f2bf(p1[r]);
        }
        asm volatile("s_wait_dscnt 0" ::: "memory");
        Frag32B pf;
        {
            const unsigned short* p = &ps[l16 * FL_LDP];
            pf.q.lo = *(const uint4*)(p + half * 8);
            pf.q.hi = *(const uint4*)(p + 16 + half * 8);
        }
        // O += P V : 4 output d-subtiles, one WMMA each (K = 32 j's).
        #pragma unroll
        for (int t = 0; t < 4; ++t) {
            Frag32B vfr;
            const unsigned short* p = &Vt[(t * 16 + l16) * FL_LDV];
            vfr.q.lo = *(const uint4*)(p + half * 8);
            vfr.q.hi = *(const uint4*)(p + 16 + half * 8);
            o[t] = __builtin_amdgcn_wmma_f32_16x16x32_bf16(
                false, pf.v, false, vfr.v, (short)0, o[t], false, false);
        }
    }

    // Epilogue: normalize, write bf16 [B][S][H] (head-major inner) for final GEMM.
    #pragma unroll
    for (int r = 0; r < 8; ++r) {
        float invl = 1.0f / lrow[r];
        int qg = q0w + half * 8 + r;
        size_t dst = ((size_t)(b * SEQ + qg)) * HID + (size_t)h * HD;
        #pragma unroll
        for (int t = 0; t < 4; ++t)
            attnb[dst + t * 16 + l16] = f2bf(o[t][r] * invl);
    }
}

// ---------------------------------------------------------------------------
// Launch
// ---------------------------------------------------------------------------
extern "C" void kernel_launch(void* const* d_in, const int* in_sizes, int n_in,
                              void* d_out, int out_size, void* d_ws, size_t ws_size,
                              hipStream_t stream)
{
    const float* x  = (const float*)d_in[0];
    const float* Wq = (const float*)d_in[1];
    const float* Wk = (const float*)d_in[2];
    const float* Wv = (const float*)d_in[3];
    const float* Wo = (const float*)d_in[4];
    float* out = (float*)d_out;

    const int M = BATCH * SEQ;   // 4096
    const int N = HID, K = HID;  // 1024

    char* ws = (char*)d_ws;
    size_t off = 0;
    auto take = [&](size_t bytes) -> char* {
        char* p = ws + off;
        off += (bytes + 255) & ~(size_t)255;
        return p;
    };
    unsigned short* xb  = (unsigned short*)take((size_t)M * K * 2);
    unsigned short* Wqb = (unsigned short*)take((size_t)N * K * 2);
    unsigned short* Wkb = (unsigned short*)take((size_t)N * K * 2);
    unsigned short* Wvb = (unsigned short*)take((size_t)N * K * 2);
    unsigned short* Wob = (unsigned short*)take((size_t)N * K * 2);
    float* qf = (float*)take((size_t)M * N * 4);
    float* kf = (float*)take((size_t)M * N * 4);
    float* vf = (float*)take((size_t)M * N * 4);
    unsigned short* qb2 = (unsigned short*)take((size_t)M * N * 2);
    unsigned short* kb2 = (unsigned short*)take((size_t)M * N * 2);
    unsigned short* vb2 = (unsigned short*)take((size_t)M * N * 2);
    unsigned short* attnb = (unsigned short*)qf;  // qf is dead after RoPE: reuse

    // 1) downconvert activations + weights
    int nx = M * K;
    k_f32_to_bf16<<<(nx + 255) / 256, 256, 0, stream>>>(x, xb, nx);
    int nw = N * K;
    k_f32_to_bf16<<<(nw + 255) / 256, 256, 0, stream>>>(Wq, Wqb, nw);
    k_f32_to_bf16<<<(nw + 255) / 256, 256, 0, stream>>>(Wk, Wkb, nw);
    k_f32_to_bf16<<<(nw + 255) / 256, 256, 0, stream>>>(Wv, Wvb, nw);
    k_f32_to_bf16<<<(nw + 255) / 256, 256, 0, stream>>>(Wo, Wob, nw);

    // 2) QKV projections (WMMA GEMM)
    dim3 gg(N / BN, M / BM);
    k_gemm_bf16<<<gg, 256, 0, stream>>>(xb, Wqb, qf, M, N, K);
    k_gemm_bf16<<<gg, 256, 0, stream>>>(xb, Wkb, kf, M, N, K);
    k_gemm_bf16<<<gg, 256, 0, stream>>>(xb, Wvb, vf, M, N, K);

    // 3) RoPE + repack to [B][NH][S][HD] bf16
    long nr = (long)BATCH * NH * SEQ * (HD / 2);
    k_rope_pack<<<(int)((nr + 255) / 256), 256, 0, stream>>>(qf, kf, vf, qb2, kb2, vb2);

    // 4) causal flash attention (WMMA)
    dim3 gf(SEQ / 64, NH, BATCH);
    k_flash<<<gf, 128, 0, stream>>>(qb2, kb2, vb2, attnb);

    // 5) output projection (WMMA GEMM) -> fp32 d_out
    k_gemm_bf16<<<gg, 256, 0, stream>>>(attnb, Wob, out, M, N, K);
}
